// SwinTransformerBlock_773094113589
// MI455X (gfx1250) — matrix operand order repaired and checked by reading, not compiled, not run on hardware
//
#include <hip/hip_runtime.h>
#include <hip/hip_bf16.h>
#include <math.h>

typedef _Float16 half_t;
typedef __attribute__((ext_vector_type(16))) _Float16 v16h;
typedef __attribute__((ext_vector_type(8)))  _Float16 v8h;
typedef __attribute__((ext_vector_type(8)))  float    v8f;

static constexpr int WSZ   = 7;
static constexpr int NN    = 49;      // tokens per window
static constexpr int CDIM  = 192;
static constexpr int HID   = 768;
static constexpr int NQKV  = 576;
static constexpr int HEADS = 6;
static constexpr int DHEAD = 32;
static constexpr int IMG   = 112;
static constexpr int NTOK  = 16 * 112 * 112;   // 200704
static constexpr int NWIN  = 4096;             // 16 * 256

// ---------------- fragment loaders (contiguous k = 16h + s layouts, wave32) ----------------
// A (16x32 f16): lane holds row (lane&15); K = kk..kk+15 contiguous,
// kk = k0 + (lane>=16 ? 16 : 0)
__device__ __forceinline__ v16h load_a16(const half_t* __restrict__ A, int lda,
                                         int m0, int k0, int lane) {
  int row = m0 + (lane & 15);
  int kk  = k0 + ((lane >> 4) << 4);
  const half_t* p = A + (size_t)row * lda + kk;
  v8h c0 = *(const v8h*)(p);
  v8h c1 = *(const v8h*)(p + 8);
  v16h a;
#pragma unroll
  for (int i = 0; i < 8; ++i) { a[i] = c0[i]; a[8 + i] = c1[i]; }
  return a;
}

// B (32x16 f16): lane holds column (lane&15); K = k0 + (lane>=16?16:0) .. +15,
// contiguous when weights are stored transposed (N-major, K contiguous).
__device__ __forceinline__ v16h load_b16(const half_t* __restrict__ Wt, int ldb,
                                         int n0, int k0, int lane) {
  int col = n0 + (lane & 15);
  int kb  = k0 + ((lane >> 4) << 4);
  const half_t* p = Wt + (size_t)col * ldb + kb;
  v8h c0 = *(const v8h*)(p);
  v8h c1 = *(const v8h*)(p + 8);
  v16h b;
#pragma unroll
  for (int i = 0; i < 8; ++i) { b[i] = c0[i]; b[8 + i] = c1[i]; }
  return b;
}

// PROJ epilogue scatter index: window row -> image-order pixel (reverse shift)
__device__ __forceinline__ size_t proj_idx(int row, int col) {
  int win = row / NN, t = row - win * NN;
  int i = t / WSZ, j = t - i * WSZ;
  int b  = win >> 8, wi = win & 255;
  int wh = wi >> 4, wv = wi & 15;
  int hh = wh * WSZ + i + 3; if (hh >= IMG) hh -= IMG;   // roll(+3)
  int ww = wv * WSZ + j + 3; if (ww >= IMG) ww -= IMG;
  return (((size_t)b * IMG + hh) * IMG + ww) * CDIM + col;
}

// ---------------- weight transpose + f16 convert ----------------
__global__ void transpose_cvt(const float* __restrict__ W, half_t* __restrict__ Wt,
                              int K, int N) {
  int o = blockIdx.x * 256 + threadIdx.x;
  if (o >= K * N) return;
  int n = o / K, k = o - n * K;
  Wt[o] = (half_t)W[(size_t)k * N + n];
}

// ---------------- LN1 + cyclic shift + window partition -> f16 ----------------
__global__ __launch_bounds__(256) void ln1_shift(const float* __restrict__ x,
                                                 const float* __restrict__ g,
                                                 const float* __restrict__ bta,
                                                 half_t* __restrict__ out) {
  int lane = threadIdx.x & 31;
  int r = blockIdx.x * 8 + (threadIdx.x >> 5);   // output window row
  int win = r / NN, t = r - win * NN;
  int i = t / WSZ, j = t - i * WSZ;
  int b  = win >> 8, wi = win & 255;
  int wh = wi >> 4, wv = wi & 15;
  int sh = wh * WSZ + i + 3; if (sh >= IMG) sh -= IMG;   // roll(-3) source
  int sw = wv * WSZ + j + 3; if (sw >= IMG) sw -= IMG;
  const float* xp = x + (((size_t)b * IMG + sh) * IMG + sw) * CDIM;
  float v[6], s = 0.f, s2 = 0.f;
#pragma unroll
  for (int tt = 0; tt < 6; ++tt) {
    float vv = xp[lane + tt * 32];
    v[tt] = vv; s += vv; s2 += vv * vv;
  }
#pragma unroll
  for (int off = 16; off > 0; off >>= 1) {
    s += __shfl_xor(s, off, 32); s2 += __shfl_xor(s2, off, 32);
  }
  float mean = s * (1.f / CDIM);
  float var  = s2 * (1.f / CDIM) - mean * mean;
  float rstd = rsqrtf(var + 1e-5f);
#pragma unroll
  for (int tt = 0; tt < 6; ++tt) {
    int c = lane + tt * 32;
    out[(size_t)r * CDIM + c] = (half_t)((v[tt] - mean) * rstd * g[c] + bta[c]);
  }
}

// ---------------- LN2 (image order) -> f16 ----------------
__global__ __launch_bounds__(256) void ln2_kernel(const float* __restrict__ x2,
                                                  const float* __restrict__ g,
                                                  const float* __restrict__ bta,
                                                  half_t* __restrict__ out) {
  int lane = threadIdx.x & 31;
  int r = blockIdx.x * 8 + (threadIdx.x >> 5);
  const float* xp = x2 + (size_t)r * CDIM;
  float v[6], s = 0.f, s2 = 0.f;
#pragma unroll
  for (int tt = 0; tt < 6; ++tt) {
    float vv = xp[lane + tt * 32];
    v[tt] = vv; s += vv; s2 += vv * vv;
  }
#pragma unroll
  for (int off = 16; off > 0; off >>= 1) {
    s += __shfl_xor(s, off, 32); s2 += __shfl_xor(s2, off, 32);
  }
  float mean = s * (1.f / CDIM);
  float var  = s2 * (1.f / CDIM) - mean * mean;
  float rstd = rsqrtf(var + 1e-5f);
#pragma unroll
  for (int tt = 0; tt < 6; ++tt) {
    int c = lane + tt * 32;
    out[(size_t)r * CDIM + c] = (half_t)((v[tt] - mean) * rstd * g[c] + bta[c]);
  }
}

// ---------------- generic WMMA GEMM: wave computes 32x64, fused epilogues ----------------
// EP: 0 = QKV (bias + scale q), 1 = PROJ (window-reverse + residual -> f32 x2),
//     2 = FC1 (bias + exact GELU -> f16), 3 = FC2 (bias + residual -> f32 out)
template <int EP>
__global__ __launch_bounds__(256) void gemm_wmma(const half_t* __restrict__ A,
                                                 const half_t* __restrict__ Wt,
                                                 const float* __restrict__ bias,
                                                 const float* __restrict__ res,
                                                 float* __restrict__ fout,
                                                 half_t* __restrict__ hout,
                                                 int M, int N, int K) {
  int lane = threadIdx.x & 31;
  int wid  = threadIdx.x >> 5;
  int m0 = (blockIdx.y * 8 + wid) * 32;
  int n0 = blockIdx.x * 64;
  if (m0 >= M) return;

  // Warm the residual lines this wave will read in the epilogue (global_prefetch_b8)
  if (EP == 1) {
    size_t idx = proj_idx(m0 + lane, n0);
    __builtin_prefetch(res + idx, 0, 0);
    __builtin_prefetch(res + idx + 32, 0, 0);
  } else if (EP == 3) {
    size_t idx = (size_t)(m0 + lane) * CDIM + n0;
    __builtin_prefetch(res + idx, 0, 0);
    __builtin_prefetch(res + idx + 32, 0, 0);
  }

  v8f acc[2][4];
#pragma unroll
  for (int mh = 0; mh < 2; ++mh)
#pragma unroll
    for (int nt = 0; nt < 4; ++nt)
#pragma unroll
      for (int e = 0; e < 8; ++e) acc[mh][nt][e] = 0.f;

  for (int k0 = 0; k0 < K; k0 += 32) {
    v16h a0 = load_a16(A, K, m0, k0, lane);
    v16h a1 = load_a16(A, K, m0 + 16, k0, lane);
#pragma unroll
    for (int nt = 0; nt < 4; ++nt) {
      v16h b = load_b16(Wt, K, n0 + nt * 16, k0, lane);
      acc[0][nt] = __builtin_amdgcn_wmma_f32_16x16x32_f16(
          false, a0, false, b, (short)0, acc[0][nt], false, false);
      acc[1][nt] = __builtin_amdgcn_wmma_f32_16x16x32_f16(
          false, a1, false, b, (short)0, acc[1][nt], false, false);
    }
  }

  int cb = lane & 15;
#pragma unroll
  for (int mh = 0; mh < 2; ++mh) {
    int rbase = m0 + mh * 16 + ((lane >> 4) << 3);
#pragma unroll
    for (int nt = 0; nt < 4; ++nt) {
      int col = n0 + nt * 16 + cb;
#pragma unroll
      for (int e = 0; e < 8; ++e) {
        int row = rbase + e;
        float val = acc[mh][nt][e] + bias[col];
        if (EP == 0) {                                     // QKV
          if (col < CDIM) val *= 0.17677669529663689f;     // 32^-0.5 on q
          hout[(size_t)row * NQKV + col] = (half_t)val;
        } else if (EP == 1) {                              // PROJ: scatter + residual
          size_t idx = proj_idx(row, col);
          fout[idx] = res[idx] + val;                      // x2 = x + attn_out
        } else if (EP == 2) {                              // FC1 + exact GELU
          val = 0.5f * val * (1.0f + erff(val * 0.70710678118654752f));
          hout[(size_t)row * HID + col] = (half_t)val;
        } else {                                           // FC2 + residual
          size_t idx = (size_t)row * CDIM + col;
          fout[idx] = res[idx] + val;
        }
      }
    }
  }
}

// ---------------- attention: one wave per (window, head) ----------------
__global__ __launch_bounds__(32) void attn_kernel(const half_t* __restrict__ qkv,
                                                  const float* __restrict__ rpb,
                                                  const float* __restrict__ amask,
                                                  half_t* __restrict__ out) {
  __shared__ float  S[64 * 64];
  __shared__ half_t P[64 * 64];
  int lane = threadIdx.x;
  int win  = blockIdx.x;
  int head = blockIdx.y;
  size_t rowbase = (size_t)win * NN;
  const half_t* qp = qkv + rowbase * NQKV + head * DHEAD;       // q channels
  const half_t* kp = qp + CDIM;                                 // k channels
  const half_t* vp = qp + 2 * CDIM;                             // v channels

  int arow = lane & 15;
  int kkA  = (lane >> 4) << 4;   // 0 / 16
  int kbB  = (lane >> 4) << 4;   // 0 / 16

  // ---- S = q @ k^T (q pre-scaled in QKV epilogue); K = d = 32, one WMMA per tile
#pragma unroll
  for (int mt = 0; mt < 4; ++mt) {
    const half_t* ap = qp + (size_t)(mt * 16 + arow) * NQKV + kkA;
    v8h a0 = *(const v8h*)(ap);
    v8h a1 = *(const v8h*)(ap + 8);
    v16h a;
#pragma unroll
    for (int i = 0; i < 8; ++i) { a[i] = a0[i]; a[8 + i] = a1[i]; }
#pragma unroll
    for (int nt = 0; nt < 4; ++nt) {
      // B[d][m] = k[m][d] -> contiguous per-lane load from k row (nt*16 + lane&15)
      const half_t* bp = kp + (size_t)(nt * 16 + arow) * NQKV + kbB;
      v8h b0 = *(const v8h*)(bp);
      v8h b1 = *(const v8h*)(bp + 8);
      v16h b;
#pragma unroll
      for (int i = 0; i < 8; ++i) { b[i] = b0[i]; b[8 + i] = b1[i]; }
      v8f c;
#pragma unroll
      for (int e = 0; e < 8; ++e) c[e] = 0.f;
      c = __builtin_amdgcn_wmma_f32_16x16x32_f16(false, a, false, b, (short)0, c,
                                                 false, false);
      int rr = mt * 16 + ((lane >> 4) << 3);
      int cc = nt * 16 + (lane & 15);
#pragma unroll
      for (int e = 0; e < 8; ++e) S[(rr + e) * 64 + cc] = c[e];
    }
  }

  // ---- preload v B-fragments NOW (independent of LDS) so the strided gathers
  //      overlap with the softmax phase below
  v16h bv[2][2];
#pragma unroll
  for (int kt = 0; kt < 2; ++kt)
#pragma unroll
    for (int nt = 0; nt < 2; ++nt) {
      int dd  = nt * 16 + (lane & 15);
      int kb0 = kt * 32 + ((lane >> 4) << 4);
      v16h b;
#pragma unroll
      for (int t = 0; t < 16; ++t)
        b[t] = vp[(size_t)(kb0 + t) * NQKV + dd];   // B[m][d] strided gather
      bv[kt][nt] = b;
    }
  __syncthreads();

  // ---- bias (computed from rpb via rel index) + mask + softmax; P = f16, padded->0
  const float* mrow = amask + (size_t)(win & 255) * (NN * NN);
#pragma unroll
  for (int hf = 0; hf < 2; ++hf) {
    int r = lane + hf * 32;
    if (r < NN) {
      int rn = r / WSZ, cn = r - rn * WSZ;
      float mx = -3.0e38f;
      for (int m = 0; m < NN; ++m) {
        int rm = m / WSZ, cm = m - rm * WSZ;
        int idx = (rn - rm + WSZ - 1) * (2 * WSZ - 1) + (cn - cm + WSZ - 1);
        float lv = S[r * 64 + m] + rpb[idx * HEADS + head] + mrow[r * NN + m];
        S[r * 64 + m] = lv;
        mx = fmaxf(mx, lv);
      }
      float sum = 0.f;
      for (int m = 0; m < NN; ++m) {
        float e = expf(S[r * 64 + m] - mx);
        S[r * 64 + m] = e;
        sum += e;
      }
      float inv = 1.0f / sum;
      for (int m = 0; m < NN; ++m) P[r * 64 + m] = (half_t)(S[r * 64 + m] * inv);
      for (int m = NN; m < 64; ++m) P[r * 64 + m] = (half_t)0.f;
    } else {
      for (int m = 0; m < 64; ++m) P[r * 64 + m] = (half_t)0.f;
    }
  }
  __syncthreads();

  // ---- O = P @ v : M=64(pad), K=64(pad, 2 steps), N=32 (2 tiles)
#pragma unroll
  for (int mt = 0; mt < 4; ++mt) {
    v8f acc[2];
#pragma unroll
    for (int nt = 0; nt < 2; ++nt)
#pragma unroll
      for (int e = 0; e < 8; ++e) acc[nt][e] = 0.f;
#pragma unroll
    for (int kt = 0; kt < 2; ++kt) {
      int row = mt * 16 + (lane & 15);
      int kk  = kt * 32 + ((lane >> 4) << 4);
      v16h a;
#pragma unroll
      for (int i = 0; i < 8; ++i) {
        a[i]     = P[row * 64 + kk + i];
        a[8 + i] = P[row * 64 + kk + 8 + i];
      }
#pragma unroll
      for (int nt = 0; nt < 2; ++nt)
        acc[nt] = __builtin_amdgcn_wmma_f32_16x16x32_f16(
            false, a, false, bv[kt][nt], (short)0, acc[nt], false, false);
    }
    int rr = mt * 16 + ((lane >> 4) << 3);
#pragma unroll
    for (int nt = 0; nt < 2; ++nt) {
      int col = head * DHEAD + nt * 16 + (lane & 15);
#pragma unroll
      for (int e = 0; e < 8; ++e) {
        int row = rr + e;
        if (row < NN)
          out[(rowbase + row) * CDIM + col] = (half_t)acc[nt][e];
      }
    }
  }
}

// ---------------- launch ----------------
extern "C" void kernel_launch(void* const* d_in, const int* in_sizes, int n_in,
                              void* d_out, int out_size, void* d_ws, size_t ws_size,
                              hipStream_t stream) {
  const float* x      = (const float*)d_in[0];
  const float* amask  = (const float*)d_in[1];
  const float* n1g    = (const float*)d_in[2];
  const float* n1b    = (const float*)d_in[3];
  const float* qkv_w  = (const float*)d_in[4];
  const float* qkv_b  = (const float*)d_in[5];
  const float* rpb    = (const float*)d_in[6];
  const float* proj_w = (const float*)d_in[7];
  const float* proj_b = (const float*)d_in[8];
  const float* n2g    = (const float*)d_in[9];
  const float* n2b    = (const float*)d_in[10];
  const float* fc1_w  = (const float*)d_in[11];
  const float* fc1_b  = (const float*)d_in[12];
  const float* fc2_w  = (const float*)d_in[13];
  const float* fc2_b  = (const float*)d_in[14];

  char* w = (char*)d_ws;
  auto alloc = [&](size_t bytes) {
    char* p = w;
    w += (bytes + 255) & ~(size_t)255;
    return p;
  };
  float*  x2   = (float*) alloc((size_t)NTOK * CDIM * 4);          // image-order f32
  half_t* tokA = (half_t*)alloc((size_t)(NTOK + 64) * CDIM * 2);   // xw -> attn_out -> y
  half_t* big  = (half_t*)alloc((size_t)NTOK * HID * 2);           // qkv (padded) / hidden
  half_t* qkvT = (half_t*)alloc((size_t)NQKV * CDIM * 2);
  half_t* projT= (half_t*)alloc((size_t)CDIM * CDIM * 2);
  half_t* fc1T = (half_t*)alloc((size_t)HID * CDIM * 2);
  half_t* fc2T = (half_t*)alloc((size_t)CDIM * HID * 2);
  (void)in_sizes; (void)n_in; (void)out_size; (void)ws_size;

  transpose_cvt<<<dim3((CDIM * NQKV + 255) / 256), 256, 0, stream>>>(qkv_w, qkvT, CDIM, NQKV);
  transpose_cvt<<<dim3((CDIM * CDIM + 255) / 256), 256, 0, stream>>>(proj_w, projT, CDIM, CDIM);
  transpose_cvt<<<dim3((CDIM * HID + 255) / 256), 256, 0, stream>>>(fc1_w, fc1T, CDIM, HID);
  transpose_cvt<<<dim3((HID * CDIM + 255) / 256), 256, 0, stream>>>(fc2_w, fc2T, HID, CDIM);

  ln1_shift<<<dim3(NTOK / 8), 256, 0, stream>>>(x, n1g, n1b, tokA);

  gemm_wmma<0><<<dim3(NQKV / 64, NTOK / 256), 256, 0, stream>>>(
      tokA, qkvT, qkv_b, nullptr, nullptr, big, NTOK, NQKV, CDIM);

  attn_kernel<<<dim3(NWIN, HEADS), 32, 0, stream>>>(big, rpb, amask, tokA);

  gemm_wmma<1><<<dim3(CDIM / 64, NTOK / 256), 256, 0, stream>>>(
      tokA, projT, proj_b, x, x2, nullptr, NTOK, CDIM, CDIM);

  ln2_kernel<<<dim3(NTOK / 8), 256, 0, stream>>>(x2, n2g, n2b, tokA);

  gemm_wmma<2><<<dim3(HID / 64, NTOK / 256), 256, 0, stream>>>(
      tokA, fc1T, fc1_b, nullptr, nullptr, big, NTOK, HID, CDIM);

  gemm_wmma<3><<<dim3(CDIM / 64, NTOK / 256), 256, 0, stream>>>(
      big, fc2T, fc2_b, x2, (float*)d_out, nullptr, NTOK, CDIM, HID);
}
